// LinearSVDBlock_1468878815667
// MI455X (gfx1250) — matrix operand order repaired and checked by reading, not compiled, not run on hardware
//
#include <hip/hip_runtime.h>
#include <cstdint>
#include <cstddef>

// ---------------------------------------------------------------- constants
static constexpr int Bc = 2, Sc = 2048, Dc = 1024;
static constexpr int Hc = 16, HDc = 64;
static constexpr int RFc = 512, INTERc = 4096;
static constexpr int Mc = Bc * Sc;              // 4096 rows
#define DEV __device__ __forceinline__

// ---------------------------------------------------------------- types
typedef __attribute__((ext_vector_type(16))) __bf16 v16bf;
typedef __attribute__((ext_vector_type(8)))  float  v8f;
typedef unsigned int u32x4 __attribute__((ext_vector_type(4)));
typedef unsigned int u32x2 __attribute__((ext_vector_type(2)));

union ABFrag { v16bf v; u32x4 q[2]; };

DEV unsigned short f2bf(float f) {
  unsigned int u = __float_as_uint(f);
  u += 0x7FFFu + ((u >> 16) & 1u);          // round-to-nearest-even
  return (unsigned short)(u >> 16);
}

// A-operand fragment (16x32 bf16): lane<16 -> row=lane, K in {0..7,16..23};
// lane>=16 -> row=lane-16, K in {8..15,24..31}.  (ISA 7.12.2)
DEV v16bf load_frag_a(const unsigned short* base, int stride, int l15, int half) {
  ABFrag f;
  const unsigned short* p = base + (size_t)l15 * stride + half * 8;
  f.q[0] = *(const u32x4*)(p);
  f.q[1] = *(const u32x4*)(p + 16);
  return f.v;
}
// B-operand fragment (32x16 bf16 = W[N=16,K=32] row-major): lane<16 -> col=lane,
// K in {0..7,16..23}; lane>=16 -> col=lane-16, K in {8..15,24..31}.
DEV v16bf load_frag_b(const unsigned short* base, int stride, int l15, int half) {
  ABFrag f;
  const unsigned short* p = base + (size_t)l15 * stride + half * 8;
  f.q[0] = *(const u32x4*)(p);
  f.q[1] = *(const u32x4*)(p + 16);
  return f.v;
}

DEV v8f wmma_bf16(v16bf a, v16bf b, v8f c) {
  return __builtin_amdgcn_wmma_f32_16x16x32_bf16(false, a, false, b, (short)0, c,
                                                 false, false);
}

// CDNA5 async global->LDS copy (ASYNCcnt-tracked; in-order completion)
DEV void async_copy_b128(const unsigned short* lds, const unsigned short* gptr) {
  unsigned ldsOff = (unsigned)(size_t)lds;           // low 32 bits = LDS offset
  unsigned long long ga = (unsigned long long)(size_t)gptr;
  asm volatile("global_load_async_to_lds_b128 %0, %1, off"
               :: "v"(ldsOff), "v"(ga) : "memory");
}
DEV void async_wait0() { asm volatile("s_wait_asynccnt 0x0" ::: "memory"); }
DEV void async_wait4() { asm volatile("s_wait_asynccnt 0x4" ::: "memory"); }

// ---------------------------------------------------------------- fp32 -> bf16
__global__ void f32_to_bf16_kernel(const float* __restrict__ x,
                                   unsigned short* __restrict__ y, size_t n4) {
  size_t i = (size_t)blockIdx.x * blockDim.x + threadIdx.x;   // index in float4s
  if (i < n4) {
    float4 v = ((const float4*)x)[i];
    u32x2 p;
    p.x = (unsigned)f2bf(v.x) | ((unsigned)f2bf(v.y) << 16);
    p.y = (unsigned)f2bf(v.z) | ((unsigned)f2bf(v.w) << 16);
    *(u32x2*)(y + i * 4) = p;
  }
}

// ---------------------------------------------------------------- layernorm
__global__ __launch_bounds__(256)
void ln_kernel(const float* __restrict__ X, const float* __restrict__ w,
               const float* __restrict__ b, unsigned short* __restrict__ Y, int D) {
  __shared__ float red[16];
  const int row = blockIdx.x;
  const int tid = threadIdx.x;
  const float* x = X + (size_t)row * D;
  float4 v = ((const float4*)x)[tid];                  // D==1024, 256 thr * 4
  float s = v.x + v.y + v.z + v.w;
  float s2 = v.x * v.x + v.y * v.y + v.z * v.z + v.w * v.w;
  for (int m = 1; m < 32; m <<= 1) { s += __shfl_xor(s, m, 32); s2 += __shfl_xor(s2, m, 32); }
  const int lane = tid & 31, wid = tid >> 5;
  if (lane == 0) { red[wid] = s; red[8 + wid] = s2; }
  __syncthreads();
  if (tid == 0) {
    float a = 0.f, c = 0.f;
    for (int i = 0; i < 8; ++i) { a += red[i]; c += red[8 + i]; }
    red[0] = a; red[8] = c;
  }
  __syncthreads();
  const float mean = red[0] / D;
  const float var  = red[8] / D - mean * mean;
  const float inv  = rsqrtf(var + 1e-5f);
  float4 ww = ((const float4*)w)[tid];
  float4 bb = ((const float4*)b)[tid];
  u32x2 p;
  p.x = (unsigned)f2bf((v.x - mean) * inv * ww.x + bb.x) |
        ((unsigned)f2bf((v.y - mean) * inv * ww.y + bb.y) << 16);
  p.y = (unsigned)f2bf((v.z - mean) * inv * ww.z + bb.z) |
        ((unsigned)f2bf((v.w - mean) * inv * ww.w + bb.w) << 16);
  *(u32x2*)(Y + (size_t)row * D + tid * 4) = p;
}

// ---------------------------------------------------------------- GEMM C = A @ W^T
// A: [M,K] bf16 ; W: [N,K] bf16.  Block tile 128x128, wave tile 32x64
// => 8 v_wmma per 32-K step.  Double-buffered async global->LDS staging:
// issue tile i+1, s_wait_asynccnt 4 (tile i landed, i+1 in flight), compute.
enum { EPI_BF16 = 0, EPI_BIAS_GELU_BF16 = 1, EPI_BIAS_RESID_F32 = 2 };

template <int EPI>
__global__ __launch_bounds__(256)
void gemm_wt_kernel(const unsigned short* __restrict__ A,
                    const unsigned short* __restrict__ W,
                    const float* __restrict__ bias,
                    const float* __restrict__ resid,
                    unsigned short* __restrict__ outBf,
                    float* __restrict__ outF,
                    int M, int N, int K) {
  // 2 x (A 128x32 + W 128x32) staging buffers; reused for epilogue transpose
  __shared__ __align__(16) unsigned short smem[16384];
  const int tid = threadIdx.x;
  const int lane = tid & 31, wid = tid >> 5;
  const int waveM = wid & 3, waveN = wid >> 2;
  const int m0 = blockIdx.x * 128;
  const int n0 = blockIdx.y * 128;
  const int l15 = lane & 15, half = lane >> 4;

  auto stageAB = [&](unsigned short* buf, int kk) {
#pragma unroll
    for (int it = 0; it < 2; ++it) {
      int c = tid + it * 256;               // chunk 0..511
      int row = c >> 2, col = (c & 3) * 8;
      async_copy_b128(buf + row * 32 + col,        A + (size_t)(m0 + row) * K + kk + col);
      async_copy_b128(buf + 4096 + row * 32 + col, W + (size_t)(n0 + row) * K + kk + col);
    }
  };

  v8f acc[2][4] = {};
  stageAB(smem, 0);

  for (int k0 = 0; k0 < K; k0 += 32) {
    unsigned short* cur = smem + ((k0 >> 5) & 1) * 8192;
    if (k0 + 32 < K) {
      stageAB(smem + (((k0 >> 5) + 1) & 1) * 8192, k0 + 32);
      async_wait4();                        // current tile complete (in-order)
    } else {
      async_wait0();
    }
    __syncthreads();

    v16bf af[2], bf[4];
#pragma unroll
    for (int i = 0; i < 2; ++i)
      af[i] = load_frag_a(cur + (waveM * 32 + i * 16) * 32, 32, l15, half);
#pragma unroll
    for (int j = 0; j < 4; ++j)
      bf[j] = load_frag_b(cur + 4096 + (waveN * 64 + j * 16) * 32, 32, l15, half);
#pragma unroll
    for (int i = 0; i < 2; ++i)
#pragma unroll
      for (int j = 0; j < 4; ++j)
        acc[i][j] = wmma_bf16(af[i], bf[j], acc[i][j]);
    __syncthreads();
  }

  if (EPI == EPI_BIAS_RESID_F32) {
    // fp32 stores coalesce to 64B per half-row: store direct
#pragma unroll
    for (int i = 0; i < 2; ++i)
#pragma unroll
      for (int j = 0; j < 4; ++j)
#pragma unroll
        for (int r = 0; r < 8; ++r) {
          int m = m0 + waveM * 32 + i * 16 + half * 8 + r;
          int n = n0 + waveN * 64 + j * 16 + l15;
          float v = acc[i][j][r] + bias[n] + resid[(size_t)m * N + n];
          outF[(size_t)m * N + n] = v;
        }
  } else {
    // bf16: transpose wave tile through LDS, then 16B-wide stores
    unsigned short* cw = smem + wid * 2048;     // 32 x 64 bf16 per wave
#pragma unroll
    for (int i = 0; i < 2; ++i)
#pragma unroll
      for (int j = 0; j < 4; ++j)
#pragma unroll
        for (int r = 0; r < 8; ++r) {
          int n = n0 + waveN * 64 + j * 16 + l15;
          float v = acc[i][j][r];
          if (EPI == EPI_BIAS_GELU_BF16) {
            v += bias[n];
            v = 0.5f * v * (1.0f + erff(v * 0.70710678118f));   // exact GELU
          }
          cw[(i * 16 + half * 8 + r) * 64 + (j * 16 + l15)] = f2bf(v);
        }
    int m = m0 + waveM * 32 + lane;             // each lane stores one row (128B)
    unsigned short* dst = outBf + (size_t)m * N + n0 + waveN * 64;
#pragma unroll
    for (int k = 0; k < 8; ++k)
      *(u32x4*)(dst + k * 8) = *(const u32x4*)(cw + lane * 64 + k * 8);
  }
}

// ---------------------------------------------------------------- per-head rank->HD
// q[b,h,s,e] = clip( t[s, h*32..+32] @ U[h]^T + bias[h] ).  K=32: one WMMA step.
// transposeOut=1 writes V as [B,H,HD,S] for the P@V stage of attention.
__global__ __launch_bounds__(256)
void head_up_kernel(const unsigned short* __restrict__ T,   // [M, H*32] bf16
                    const unsigned short* __restrict__ U,   // [H,64,32] bf16
                    const float* __restrict__ bias,          // [H,64]
                    unsigned short* __restrict__ out,
                    int transposeOut) {
  __shared__ __align__(16) unsigned short hb[8][1024];      // per-wave 16x64 / 64x16
  const int tid = threadIdx.x, lane = tid & 31, wid = tid >> 5;
  const int h = blockIdx.y;
  const int m0 = blockIdx.x * 128 + wid * 16;
  const int l15 = lane & 15, half = lane >> 4;
  const int b_ = m0 / Sc, s0 = m0 % Sc;

  v16bf a = load_frag_a(T + (size_t)m0 * (Hc * 32) + h * 32, Hc * 32, l15, half);
  v8f acc[4] = {};
#pragma unroll
  for (int t = 0; t < 4; ++t) {
    v16bf b = load_frag_b(U + ((size_t)h * 64 + t * 16) * 32, 32, l15, half);
    acc[t] = wmma_bf16(a, b, acc[t]);
  }
  if (!transposeOut) {
    // tile layout [s(16)][e(64)]
#pragma unroll
    for (int t = 0; t < 4; ++t)
#pragma unroll
      for (int r = 0; r < 8; ++r) {
        float v = acc[t][r] + bias[h * 64 + t * 16 + l15];
        v = fminf(fmaxf(v, -1.0e6f), 1.0e6f);
        hb[wid][(half * 8 + r) * 64 + t * 16 + l15] = f2bf(v);
      }
    // lane stores half a row (64B) of [B,H,S,64]
    unsigned short* dst = out + (((size_t)(b_ * Hc + h) * Sc + s0 + l15) * HDc) + half * 32;
#pragma unroll
    for (int k = 0; k < 4; ++k)
      *(u32x4*)(dst + k * 8) = *(const u32x4*)(&hb[wid][l15 * 64 + half * 32 + k * 8]);
  } else {
    // tile layout [e(64)][s(16)]
#pragma unroll
    for (int t = 0; t < 4; ++t)
#pragma unroll
      for (int r = 0; r < 8; ++r) {
        float v = acc[t][r] + bias[h * 64 + t * 16 + l15];
        v = fminf(fmaxf(v, -1.0e6f), 1.0e6f);
        hb[wid][(t * 16 + l15) * 16 + half * 8 + r] = f2bf(v);
      }
    // lane stores two rows (e) of 16 s-values each into [B,H,64,S]
#pragma unroll
    for (int d = 0; d < 2; ++d) {
      int e = lane * 2 + d;
      unsigned short* dst = out + ((size_t)(b_ * Hc + h) * HDc + e) * Sc + s0;
#pragma unroll
      for (int k = 0; k < 2; ++k)
        *(u32x4*)(dst + k * 8) = *(const u32x4*)(&hb[wid][e * 16 + k * 8]);
    }
  }
}

// ---------------------------------------------------------------- flash attention
// Block = 128 query rows (8 waves x 16).  K/V tiles staged once per block in
// LDS via async copies and shared by all waves; waves past their causal limit
// skip compute via a wave-uniform branch (EXEC stays all-1 for WMMA).
__global__ __launch_bounds__(256)
void attn_kernel(const unsigned short* __restrict__ Q,    // [B,H,S,64]
                 const unsigned short* __restrict__ Kc,   // [B,H,S,64]
                 const unsigned short* __restrict__ Vt,   // [B,H,64,S]
                 unsigned short* __restrict__ O) {        // [B,S,H*64]
  __shared__ __align__(16) unsigned short smem[8192];     // kt(2K) vt(2K) pbuf(4K)
  unsigned short* kt = smem;                 // 32 keys x 64
  unsigned short* vt = smem + 2048;          // 64 e    x 32 keys
  const int tid = threadIdx.x, lane = tid & 31, wid = tid >> 5;
  unsigned short* pb = smem + 4096 + wid * 512;   // per-wave 16 x 32
  const int bh = blockIdx.y;                               // b*H + h
  const int q0 = blockIdx.x * 128 + wid * 16;
  const int jEnd = blockIdx.x * 128 + 127;                 // top wave's causal limit
  const int l15 = lane & 15, half = lane >> 4;
  const size_t base = (size_t)bh * Sc * HDc;

  v16bf qa[2];
#pragma unroll
  for (int f = 0; f < 2; ++f)
    qa[f] = load_frag_a(Q + base + (size_t)q0 * HDc + f * 32, HDc, l15, half);

  float mrow[8], lrow[8];
#pragma unroll
  for (int r = 0; r < 8; ++r) { mrow[r] = -3.0e38f; lrow[r] = 0.f; }
  v8f o[4] = {};
  const float scale = 0.125f;                              // 1/sqrt(64)

  for (int j0 = 0; j0 <= jEnd; j0 += 32) {
    // stage K tile (rows j0..j0+31 x 64) and V^T tile (64 x keys j0..j0+31)
    {
      int krow = tid >> 3, kcol = (tid & 7) * 8;
      async_copy_b128(kt + krow * 64 + kcol, Kc + base + (size_t)(j0 + krow) * HDc + kcol);
      int vrow = tid >> 2, vcol = (tid & 3) * 8;
      async_copy_b128(vt + vrow * 32 + vcol, Vt + ((size_t)bh * HDc + vrow) * Sc + j0 + vcol);
    }
    async_wait0();
    __syncthreads();

    if (j0 <= q0 + 15) {                     // wave-uniform causal guard
      v8f s[2] = {};
#pragma unroll
      for (int nh = 0; nh < 2; ++nh)
#pragma unroll
        for (int f = 0; f < 2; ++f) {
          v16bf kf = load_frag_b(kt + (nh * 16) * 64 + f * 32, 64, l15, half);
          s[nh] = wmma_bf16(qa[f], kf, s[nh]);
        }
      const bool diag = (j0 + 31 > q0);
#pragma unroll
      for (int nh = 0; nh < 2; ++nh)
#pragma unroll
        for (int r = 0; r < 8; ++r) {
          float v = s[nh][r] * scale;
          if (diag) {
            int key = j0 + nh * 16 + l15;
            int qq  = q0 + half * 8 + r;
            if (key > qq) v = -3.0e38f;
          }
          s[nh][r] = v;
        }
      float pw[2][8];
#pragma unroll
      for (int r = 0; r < 8; ++r) {
        float rm = fmaxf(s[0][r], s[1][r]);
        rm = fmaxf(rm, __shfl_xor(rm, 1, 32));
        rm = fmaxf(rm, __shfl_xor(rm, 2, 32));
        rm = fmaxf(rm, __shfl_xor(rm, 4, 32));
        rm = fmaxf(rm, __shfl_xor(rm, 8, 32));
        float mnew = fmaxf(mrow[r], rm);
        float corr = __expf(mrow[r] - mnew);
        float p0 = __expf(s[0][r] - mnew);
        float p1 = __expf(s[1][r] - mnew);
        float rs = p0 + p1;
        rs += __shfl_xor(rs, 1, 32);
        rs += __shfl_xor(rs, 2, 32);
        rs += __shfl_xor(rs, 4, 32);
        rs += __shfl_xor(rs, 8, 32);
        lrow[r] = lrow[r] * corr + rs;
        mrow[r] = mnew;
        pw[0][r] = p0; pw[1][r] = p1;
#pragma unroll
        for (int t = 0; t < 4; ++t) o[t][r] *= corr;
      }
      // C-layout -> A-layout for P through per-wave LDS scratch
#pragma unroll
      for (int nh = 0; nh < 2; ++nh)
#pragma unroll
        for (int r = 0; r < 8; ++r)
          pb[(half * 8 + r) * 32 + nh * 16 + l15] = f2bf(pw[nh][r]);
      v16bf pf = load_frag_a(pb, 32, l15, half);
#pragma unroll
      for (int t = 0; t < 4; ++t) {
        v16bf vf = load_frag_b(vt + (t * 16) * 32, 32, l15, half);
        o[t] = wmma_bf16(pf, vf, o[t]);
      }
    }
    __syncthreads();
  }

  // normalize, transpose through LDS (smem reused), wide stores to [B,S,H*64]
  __syncthreads();
  unsigned short* ob = smem + wid * 1024;   // 16 x 64 per wave
  const int b_ = bh / Hc, h = bh % Hc;
#pragma unroll
  for (int t = 0; t < 4; ++t)
#pragma unroll
    for (int r = 0; r < 8; ++r)
      ob[(half * 8 + r) * 64 + t * 16 + l15] = f2bf(o[t][r] / lrow[r]);
  unsigned short* dst = O + ((size_t)b_ * Sc + q0 + l15) * Dc + h * HDc + half * 32;
#pragma unroll
  for (int k = 0; k < 4; ++k)
    *(u32x4*)(dst + k * 8) = *(const u32x4*)(ob + l15 * 64 + half * 32 + k * 8);
}

// ---------------------------------------------------------------- launcher
extern "C" void kernel_launch(void* const* d_in, const int* in_sizes, int n_in,
                              void* d_out, int out_size, void* d_ws, size_t ws_size,
                              hipStream_t stream) {
  (void)in_sizes; (void)n_in; (void)out_size; (void)ws_size;
  const float* hidden   = (const float*)d_in[0];
  const float* q_U      = (const float*)d_in[1];
  const float* q_V      = (const float*)d_in[2];
  const float* q_bias   = (const float*)d_in[3];
  const float* k_U      = (const float*)d_in[4];
  const float* k_V      = (const float*)d_in[5];
  const float* k_bias   = (const float*)d_in[6];
  const float* v_U      = (const float*)d_in[7];
  const float* v_V      = (const float*)d_in[8];
  const float* v_bias   = (const float*)d_in[9];
  const float* out_U    = (const float*)d_in[10];
  const float* out_V    = (const float*)d_in[11];
  const float* out_bias = (const float*)d_in[12];
  const float* fc1_U    = (const float*)d_in[13];
  const float* fc1_V    = (const float*)d_in[14];
  const float* fc1_bias = (const float*)d_in[15];
  const float* fc2_U    = (const float*)d_in[16];
  const float* fc2_V    = (const float*)d_in[17];
  const float* fc2_bias = (const float*)d_in[18];
  const float* ln1_w    = (const float*)d_in[19];
  const float* ln1_b    = (const float*)d_in[20];
  const float* ln2_w    = (const float*)d_in[21];
  const float* ln2_b    = (const float*)d_in[22];
  float* out = (float*)d_out;

  char* ws = (char*)d_ws;
  size_t off = 0;
  auto alloc = [&](size_t bytes) -> char* {
    char* p = ws + off;
    off += (bytes + 255) & ~(size_t)255;
    return p;
  };
  typedef unsigned short bf;
  const size_t VW = (size_t)RFc * Dc;        // 512*1024 elems
  const size_t UW = (size_t)Hc * HDc * 32;   // per-head U
  bf* wqV = (bf*)alloc(VW * 2);  bf* wkV = (bf*)alloc(VW * 2);  bf* wvV = (bf*)alloc(VW * 2);
  bf* wqU = (bf*)alloc(UW * 2);  bf* wkU = (bf*)alloc(UW * 2);  bf* wvU = (bf*)alloc(UW * 2);
  bf* woV = (bf*)alloc(VW * 2);                       // out_V [512,1024]
  bf* woU = (bf*)alloc((size_t)Dc * 512 * 2);         // out_U [1024,512]
  bf* w1V = (bf*)alloc(VW * 2);                       // fc1_V [512,1024]
  bf* w1U = (bf*)alloc((size_t)INTERc * RFc * 2);     // fc1_U [4096,512]
  bf* w2V = (bf*)alloc((size_t)RFc * INTERc * 2);     // fc2_V [512,4096]
  bf* w2U = (bf*)alloc((size_t)Dc * RFc * 2);         // fc2_U [1024,512]
  bf* x1   = (bf*)alloc((size_t)Mc * Dc * 2);
  bf* tbuf = (bf*)alloc((size_t)Mc * 512 * 2);
  bf* qb   = (bf*)alloc((size_t)Mc * Dc * 2);
  bf* kb   = (bf*)alloc((size_t)Mc * Dc * 2);
  bf* vT   = (bf*)alloc((size_t)Mc * Dc * 2);
  bf* attn = (bf*)alloc((size_t)Mc * Dc * 2);
  bf* o1   = (bf*)alloc((size_t)Mc * 512 * 2);
  float* hbuf = (float*)alloc((size_t)Mc * Dc * 4);
  bf* n2   = (bf*)alloc((size_t)Mc * Dc * 2);
  bf* fft  = (bf*)alloc((size_t)Mc * 512 * 2);
  bf* ffb  = (bf*)alloc((size_t)Mc * INTERc * 2);
  bf* gb   = (bf*)alloc((size_t)Mc * 512 * 2);

  // ---- weight conversions (fp32 -> bf16), vectorized x4
  struct CV { const float* s; bf* d; size_t n; } cv[] = {
      {q_V, wqV, VW}, {k_V, wkV, VW}, {v_V, wvV, VW},
      {q_U, wqU, UW}, {k_U, wkU, UW}, {v_U, wvU, UW},
      {out_V, woV, VW}, {out_U, woU, (size_t)Dc * 512},
      {fc1_V, w1V, VW}, {fc1_U, w1U, (size_t)INTERc * RFc},
      {fc2_V, w2V, (size_t)RFc * INTERc}, {fc2_U, w2U, (size_t)Dc * RFc},
  };
  for (auto& c : cv) {
    size_t n4 = c.n / 4;
    f32_to_bf16_kernel<<<dim3((unsigned)((n4 + 255) / 256)), 256, 0, stream>>>(c.s, c.d, n4);
  }

  // ---- LN1
  ln_kernel<<<Mc, 256, 0, stream>>>(hidden, ln1_w, ln1_b, x1, Dc);

  const dim3 gN512(Mc / 128, 512 / 128), gN1024(Mc / 128, Dc / 128), gN4096(Mc / 128, INTERc / 128);
  const dim3 gHead(Mc / 128, Hc);

  // ---- QKV (low-rank, two WMMA stages each)
  gemm_wt_kernel<EPI_BF16><<<gN512, 256, 0, stream>>>(x1, wqV, nullptr, nullptr, tbuf, nullptr, Mc, 512, Dc);
  head_up_kernel<<<gHead, 256, 0, stream>>>(tbuf, wqU, q_bias, qb, 0);
  gemm_wt_kernel<EPI_BF16><<<gN512, 256, 0, stream>>>(x1, wkV, nullptr, nullptr, tbuf, nullptr, Mc, 512, Dc);
  head_up_kernel<<<gHead, 256, 0, stream>>>(tbuf, wkU, k_bias, kb, 0);
  gemm_wt_kernel<EPI_BF16><<<gN512, 256, 0, stream>>>(x1, wvV, nullptr, nullptr, tbuf, nullptr, Mc, 512, Dc);
  head_up_kernel<<<gHead, 256, 0, stream>>>(tbuf, wvU, v_bias, vT, 1);   // V stored transposed

  // ---- causal flash attention
  attn_kernel<<<dim3(Sc / 128, Bc * Hc), 256, 0, stream>>>(qb, kb, vT, attn);

  // ---- low-rank out projection + residual (h = hidden + attn_proj)
  gemm_wt_kernel<EPI_BF16><<<gN512, 256, 0, stream>>>(attn, woV, nullptr, nullptr, o1, nullptr, Mc, 512, Dc);
  gemm_wt_kernel<EPI_BIAS_RESID_F32><<<gN1024, 256, 0, stream>>>(o1, woU, out_bias, hidden, nullptr, hbuf, Mc, Dc, 512);

  // ---- LN2 + low-rank FFN with exact GELU
  ln_kernel<<<Mc, 256, 0, stream>>>(hbuf, ln2_w, ln2_b, n2, Dc);
  gemm_wt_kernel<EPI_BF16><<<gN512, 256, 0, stream>>>(n2, w1V, nullptr, nullptr, fft, nullptr, Mc, 512, Dc);
  gemm_wt_kernel<EPI_BIAS_GELU_BF16><<<gN4096, 256, 0, stream>>>(fft, w1U, fc1_bias, nullptr, ffb, nullptr, Mc, INTERc, RFc);
  gemm_wt_kernel<EPI_BF16><<<gN512, 256, 0, stream>>>(ffb, w2V, nullptr, nullptr, gb, nullptr, Mc, 512, INTERc);
  // final: d_out = h + ff2
  gemm_wt_kernel<EPI_BIAS_RESID_F32><<<gN1024, 256, 0, stream>>>(gb, w2U, fc2_bias, hbuf, nullptr, out, Mc, Dc, RFc);
}
